// ScaledDotProductAttention_29454885716146
// MI455X (gfx1250) — matrix operand order repaired and checked by reading, not compiled, not run on hardware
//
#include <hip/hip_runtime.h>
#include <hip/hip_bf16.h>
#include <math.h>

// ---------------------------------------------------------------------------
// Softmax-over-query attention (B=4, S=4096, D=64, fp32 io) for gfx1250.
//   scores = (Q @ K^T) * 1/8
//   attn   = softmax(scores, axis=q)        <- per-KEY-column normalization
//   out    = attn @ V
// Two kernels:
//   1) stats_kernel : per (b,k) column max m and inv-sum 1/den  (WMMA GEMM)
//   2) attn_kernel  : out = exp(s - m[k]) @ (V * invden[k])     (2x WMMA GEMM)
// All matmuls use v_wmma_f32_16x16x32_f16 (f16 operands, f32 accumulate).
// ---------------------------------------------------------------------------

typedef __attribute__((ext_vector_type(16))) _Float16 v16h;
typedef __attribute__((ext_vector_type(8)))  _Float16 v8h;
typedef __attribute__((ext_vector_type(8)))  float    v8f;

#define WMMA_F16(A, B, C) \
  __builtin_amdgcn_wmma_f32_16x16x32_f16(false, (A), false, (B), (short)0, (C), false, false)

static __device__ __forceinline__ v16h cat8(v8h lo, v8h hi) {
  return __builtin_shufflevector(lo, hi, 0,1,2,3,4,5,6,7,8,9,10,11,12,13,14,15);
}

// A fragment (16x32 f16, M=row, Kdim=col) from row-major LDS [rows][ld].
// Lane half h holds K = 16h+0..15 contiguous (matches load_b).
static __device__ __forceinline__ v16h load_a(const _Float16* base, int row, int ld,
                                              int kchunk, int half) {
  const _Float16* p = base + row * ld + kchunk * 32 + 16 * half;
  v8h lo = *(const v8h*)(p);
  v8h hi = *(const v8h*)(p + 8);
  return cat8(lo, hi);
}

// B fragment (32x16 f16, Kdim x N). We store B^T row-major: mat[col][k], so each
// lane (col = lane&15, half h) reads 16 contiguous K halves at k = 16h (32B aligned).
static __device__ __forceinline__ v16h load_b(const _Float16* base, int col, int ld,
                                              int kchunk, int half) {
  const _Float16* p = base + col * ld + kchunk * 32 + 16 * half;
  return *(const v16h*)(p);
}

// ---------------------------------------------------------------------------
// Kernel 1: per-key-column softmax stats over the query axis.
//   m[b,k]      = max_q (Q*0.125) . K[k]
//   invden[b,k] = 1 / sum_q exp(s - m)
// Grid: (S/128, B). 8 waves x 16 k-columns each; loop over q in 64-row chunks.
// C fragment has N = k = lane&15 fixed per lane, M = q values -> reduce over the
// 8 in-lane floats online, merge lane L with L+16 (same column) at the end.
// ---------------------------------------------------------------------------
__global__ __launch_bounds__(256)
void stats_kernel(const float* __restrict__ Q, const float* __restrict__ K,
                  float* __restrict__ m_ws, float* __restrict__ id_ws) {
  constexpr int S = 4096, D = 64;
  constexpr int KT = 128;   // k columns per workgroup
  constexpr int QC = 64;    // q rows staged per iteration
  __shared__ __align__(128) _Float16 Ks[KT * D];   // 16 KB
  __shared__ __align__(128) _Float16 Qs[QC * D];   //  8 KB

  const int tid  = threadIdx.x;
  const int lane = tid & 31, wid = tid >> 5;
  const int half = lane >> 4, l16 = lane & 15;
  const int b  = blockIdx.y;
  const int k0 = blockIdx.x * KT;

  const float* Kg = K + ((size_t)b * S + k0) * D;
  for (int i = tid; i < KT * D; i += 256) Ks[i] = (_Float16)Kg[i];
  __syncthreads();

  // This wave's 16 key columns: B fragments are loop-invariant.
  v16h bk0 = load_b(Ks, wid * 16 + l16, D, 0, half);
  v16h bk1 = load_b(Ks, wid * 16 + l16, D, 1, half);

  float m = -INFINITY, ssum = 0.0f;

  for (int qc = 0; qc < S; qc += QC) {
    __syncthreads();
    const float* Qg = Q + ((size_t)b * S + qc) * D;
    for (int i = tid; i < QC * D; i += 256)
      Qs[i] = (_Float16)(Qg[i] * 0.125f);       // fold 1/sqrt(64), exact in f16
    __syncthreads();

    #pragma unroll
    for (int ms = 0; ms < QC / 16; ++ms) {
      v16h a0 = load_a(Qs, ms * 16 + l16, D, 0, half);
      v16h a1 = load_a(Qs, ms * 16 + l16, D, 1, half);
      v8f c = {};
      c = WMMA_F16(a0, bk0, c);
      c = WMMA_F16(a1, bk1, c);
      float vmax = c[0];
      #pragma unroll
      for (int r = 1; r < 8; ++r) vmax = fmaxf(vmax, c[r]);
      float nm = fmaxf(m, vmax);
      float ps = 0.0f;
      #pragma unroll
      for (int r = 0; r < 8; ++r) ps += __expf(c[r] - nm);
      ssum = ssum * __expf(m - nm) + ps;
      m = nm;
    }
  }

  // Merge the two q-halves of each column (lane L <-> L+16).
  float mo = __shfl_xor(m, 16, 32);
  float so = __shfl_xor(ssum, 16, 32);
  float mf = fmaxf(m, mo);
  float sf = ssum * __expf(m - mf) + so * __expf(mo - mf);
  if (half == 0) {
    size_t idx = (size_t)b * S + k0 + wid * 16 + l16;
    m_ws[idx]  = mf;
    id_ws[idx] = 1.0f / sf;
  }
}

// ---------------------------------------------------------------------------
// Kernel 2: out[b,q,:] = sum_k exp(s[q,k] - m[k]) * (V[k,:] * invden[k])
// Grid: (S/128, B). Each wave owns a 16-q strip and all 64 output columns
// (4 C fragments). Per 64-key chunk: WMMA scores, exp in-register, stage P
// through a per-wave LDS slab to re-fragment as an A matrix, WMMA into out.
// ---------------------------------------------------------------------------
__global__ __launch_bounds__(256)
void attn_kernel(const float* __restrict__ Q, const float* __restrict__ K,
                 const float* __restrict__ V, const float* __restrict__ m_ws,
                 const float* __restrict__ id_ws, float* __restrict__ Out) {
  constexpr int S = 4096, D = 64;
  constexpr int QT = 128;   // q rows per workgroup
  constexpr int KC = 64;    // keys per chunk
  __shared__ __align__(128) _Float16 Qs [QT * D];      // 16 KB, [q][d], scaled
  __shared__ __align__(128) _Float16 Ks [KC * D];      //  8 KB, [k][d]
  __shared__ __align__(128) _Float16 VsT[D * KC];      //  8 KB, [d][k], * invden
  __shared__ __align__(128) _Float16 Ps [8 * 16 * KC]; // 16 KB, per-wave P tiles
  __shared__ float mcol[KC];

  const int tid  = threadIdx.x;
  const int lane = tid & 31, wid = tid >> 5;
  const int half = lane >> 4, l16 = lane & 15;
  const int b  = blockIdx.y;
  const int q0 = blockIdx.x * QT;

  const float* Qg = Q + ((size_t)b * S + q0) * D;
  for (int i = tid; i < QT * D; i += 256)
    Qs[i] = (_Float16)(Qg[i] * 0.125f);   // identical conversion to stats pass
  __syncthreads();

  // This wave's Q fragments (rows wid*16..+15) are loop-invariant.
  v16h aq0 = load_a(Qs, wid * 16 + l16, D, 0, half);
  v16h aq1 = load_a(Qs, wid * 16 + l16, D, 1, half);

  v8f acc[4] = {};
  _Float16* Pw = Ps + wid * 16 * KC;

  for (int kc = 0; kc < S; kc += KC) {
    __syncthreads();
    const float* Kg = K + ((size_t)b * S + kc) * D;
    const float* Vg = V + ((size_t)b * S + kc) * D;
    for (int i = tid; i < KC * D; i += 256) {
      Ks[i] = (_Float16)Kg[i];
      int kk = i / D, d = i % D;
      VsT[d * KC + kk] = (_Float16)(Vg[i] * id_ws[(size_t)b * S + kc + kk]);
    }
    if (tid < KC) mcol[tid] = m_ws[(size_t)b * S + kc + tid];
    __syncthreads();

    // GEMM1 + exp: P[16 x KC] in per-wave LDS slab (wave-private, no barrier).
    #pragma unroll
    for (int ks = 0; ks < KC / 16; ++ks) {
      v16h b0 = load_b(Ks, ks * 16 + l16, D, 0, half);
      v16h b1 = load_b(Ks, ks * 16 + l16, D, 1, half);
      v8f s = {};
      s = WMMA_F16(aq0, b0, s);
      s = WMMA_F16(aq1, b1, s);
      float mk = mcol[ks * 16 + l16];   // N = lane&15: column max, lane-uniform
      #pragma unroll
      for (int r = 0; r < 8; ++r)
        Pw[(r + 8 * half) * KC + ks * 16 + l16] = (_Float16)__expf(s[r] - mk);
    }

    // GEMM2: acc(16x64) += P(16xKC) @ Vhat(KCx64)
    #pragma unroll
    for (int kk2 = 0; kk2 < KC / 32; ++kk2) {
      v16h ap = load_a(Pw, l16, KC, kk2, half);
      #pragma unroll
      for (int dt = 0; dt < 4; ++dt) {
        v16h bv = load_b(VsT, dt * 16 + l16, KC, kk2, half);
        acc[dt] = WMMA_F16(ap, bv, acc[dt]);
      }
    }
  }

  float* Og = Out + ((size_t)b * S + q0) * D;
  #pragma unroll
  for (int dt = 0; dt < 4; ++dt)
    #pragma unroll
    for (int r = 0; r < 8; ++r)
      Og[(wid * 16 + r + 8 * half) * D + dt * 16 + l16] = acc[dt][r];
}

// ---------------------------------------------------------------------------
extern "C" void kernel_launch(void* const* d_in, const int* in_sizes, int n_in,
                              void* d_out, int out_size, void* d_ws, size_t ws_size,
                              hipStream_t stream) {
  constexpr int B = 4, S = 4096;
  const float* Q = (const float*)d_in[0];
  const float* K = (const float*)d_in[1];
  const float* V = (const float*)d_in[2];
  float* Out = (float*)d_out;

  float* m_ws  = (float*)d_ws;           // B*S floats
  float* id_ws = m_ws + (size_t)B * S;   // B*S floats (128 KB total)

  dim3 grid(S / 128, B);
  stats_kernel<<<grid, 256, 0, stream>>>(Q, K, m_ws, id_ws);
  attn_kernel <<<grid, 256, 0, stream>>>(Q, K, V, m_ws, id_ws, Out);
}
